// W8X8Linear_18502719111671
// MI455X (gfx1250) — 8-bit matrix operand order repaired and checked by reading, compiled, not run on hardware
//
#include <hip/hip_runtime.h>

typedef int v8i __attribute__((ext_vector_type(8)));
typedef int v4i __attribute__((ext_vector_type(4)));

__global__ __launch_bounds__(256)
void quant_rows_1024(const float* __restrict__ x,
                     signed char* __restrict__ q,
                     float* __restrict__ rowmax)
{
    const int tid  = threadIdx.x;
    const int lane = tid & 31;
    const int wave = tid >> 5;
    const size_t row = blockIdx.x;

    const float4 v = ((const float4*)(x + row * 1024))[tid];
    float m = fmaxf(fmaxf(fabsf(v.x), fabsf(v.y)),
                    fmaxf(fabsf(v.z), fabsf(v.w)));

#pragma unroll
    for (int off = 16; off > 0; off >>= 1)
        m = fmaxf(m, __shfl_xor(m, off, 32));

    __shared__ float sm[8];
    if (lane == 0) sm[wave] = m;
    __syncthreads();

    float rm = sm[0];
#pragma unroll
    for (int i = 1; i < 8; ++i) rm = fmaxf(rm, sm[i]);

    const float s = 127.0f / rm;
    const int c0 = (int)(signed char)(v.x * s);
    const int c1 = (int)(signed char)(v.y * s);
    const int c2 = (int)(signed char)(v.z * s);
    const int c3 = (int)(signed char)(v.w * s);
    const int packed = (c0 & 0xff) | ((c1 & 0xff) << 8) |
                       ((c2 & 0xff) << 16) | ((c3 & 0xff) << 24);
    ((int*)q)[row * 256 + tid] = packed;

    if (tid == 0) rowmax[row] = rm;
}

__global__ __launch_bounds__(256)
void w8a8_wmma_gemm(const signed char* __restrict__ xq,
                    const float* __restrict__ xmax,
                    const signed char* __restrict__ wq,
                    const float* __restrict__ wmax,
                    float* __restrict__ out,
                    int M, int N, int K)
{
    const int tid  = threadIdx.x;
    const int lane = tid & 31;
    const int wave = tid >> 5;
    const int l16  = lane & 15;
    const int half = lane >> 4;

    const int m0 = blockIdx.y * 128 + wave * 16;
    const int n0 = blockIdx.x * 64;

    union AV { v8i v; unsigned long long d[4]; };
    union BV { v8i v; v4i q[2]; unsigned long long d[4]; };

    v8i acc[4] = {};

    const signed char* arow = xq + (size_t)(m0 + l16) * K + (half << 3);
    const signed char* bcol = wq + (size_t)(n0 + l16) * K + (half << 3);

#define LOAD_A(A, kk)                                                         \
    do {                                                                      \
        const unsigned long long* ap =                                        \
            (const unsigned long long*)(arow + (kk));                         \
        (A).d[0] = ap[0]; (A).d[1] = ap[2];                                   \
        (A).d[2] = ap[4]; (A).d[3] = ap[6];                                   \
    } while (0)

#define LOAD_B(B, t, kk)                                                      \
    do {                                                                      \
        const unsigned long long* bp =                                        \
            (const unsigned long long*)(bcol + (size_t)(t) * 16 * K + (kk));  \
        (B).d[0] = bp[0]; (B).d[1] = bp[2];                                   \
        (B).d[2] = bp[4]; (B).d[3] = bp[6];                                   \
    } while (0)

#define WMMA4(A, B)                                                           \
    do {                                                                      \
        _Pragma("unroll")                                                     \
        for (int t = 0; t < 4; ++t)                                           \
            acc[t] = __builtin_amdgcn_wmma_i32_16x16x64_iu8(                  \
                         true, (A).v, true, (B)[t].v, acc[t], false, false);  \
    } while (0)

    AV a0, a1;
    BV b0[4], b1[4];

    LOAD_A(a0, 0);
#pragma unroll
    for (int t = 0; t < 4; ++t) LOAD_B(b0[t], t, 0);

    int k = 0;
    for (; k + 128 < K; k += 128) {
        LOAD_A(a1, k + 64);
#pragma unroll
        for (int t = 0; t < 4; ++t) LOAD_B(b1[t], t, k + 64);
        WMMA4(a0, b0);

        LOAD_A(a0, k + 128);
#pragma unroll
        for (int t = 0; t < 4; ++t) LOAD_B(b0[t], t, k + 128);
        WMMA4(a1, b1);
    }
    LOAD_A(a1, k + 64);
#pragma unroll
    for (int t = 0; t < 4; ++t) LOAD_B(b1[t], t, k + 64);
    WMMA4(a0, b0);
    WMMA4(a1, b1);

#undef LOAD_A
#undef LOAD_B
#undef WMMA4

    const float inv = 1.0f / (127.0f * 127.0f);
    const int mbase = m0 + (half << 3);
#pragma unroll
    for (int t = 0; t < 4; ++t) {
        const int n = n0 + t * 16 + l16;
        const float wm = wmax[n];
#pragma unroll
        for (int r = 0; r < 8; ++r) {
            const int m = mbase + r;
            const float dq = (float)acc[t][r] * inv;
            const _Float16 h = (_Float16)dq;
            out[(size_t)m * N + n] = (float)h * (xmax[m] * wm);
        }
    }
}

extern "C" void kernel_launch(void* const* d_in, const int* in_sizes, int n_in,
                              void* d_out, int out_size, void* d_ws, size_t ws_size,
                              hipStream_t stream)
{
    const float* x = (const float*)d_in[0];
    const float* w = (const float*)d_in[1];
    float* out = (float*)d_out;

    const int K = 1024;
    const int N = 1024;
    const int M = in_sizes[0] / K;

    char* ws = (char*)d_ws;
    size_t off = 0;
    signed char* xq = (signed char*)(ws + off); off += (size_t)M * K;
    signed char* wq = (signed char*)(ws + off); off += (size_t)N * K;
    off = (off + 255) & ~(size_t)255;
    float* xmax = (float*)(ws + off);           off += (size_t)M * sizeof(float);
    float* wmax = (float*)(ws + off);           off += (size_t)N * sizeof(float);

    quant_rows_1024<<<N, 256, 0, stream>>>(w, wq, wmax);
    quant_rows_1024<<<M, 256, 0, stream>>>(x, xq, xmax);

    dim3 grid(N / 64, M / 128);
    w8a8_wmma_gemm<<<grid, 256, 0, stream>>>(xq, xmax, wq, wmax, out, M, N, K);
}
